// EmbedGIN_2104533975646
// MI455X (gfx1250) — matrix operand order repaired and checked by reading, not compiled, not run on hardware
//
#include <hip/hip_runtime.h>
#include <hip/hip_bf16.h>
#include <cstdint>

#define N_NODES 100000
#define N_EDGES 600000
#define H 128
#define LAYERS 4
#define ATYPES 100
#define OUTDIM 10
#define GRAPHS 128
#define BN_EPS 1e-5f

typedef __attribute__((ext_vector_type(16))) __bf16 v16bf;
typedef __attribute__((ext_vector_type(8)))  float  v8f;
typedef unsigned int u32;
typedef __attribute__((ext_vector_type(4))) u32 v4u;
typedef __attribute__((ext_vector_type(8))) int v8i;
typedef __attribute__((ext_vector_type(4))) int v4i;

// ---------- fold BN(+bias) into per-column scale/shift (9 slots: 4x bn1, 4x bn2, lin1) ----------
__global__ void k_prep(const float* b1, const float* g1, const float* be1, const float* m1, const float* v1,
                       const float* b2, const float* g2, const float* be2, const float* m2, const float* v2,
                       const float* lin1_b, float* scale, float* shift) {
  int t = blockIdx.x * blockDim.x + threadIdx.x;
  if (t >= 9 * H) return;
  int slot = t / H, j = t % H;
  if (slot < 4) {
    int idx = slot * H + j;
    float sc = g1[idx] * rsqrtf(v1[idx] + BN_EPS);
    scale[t] = sc;
    shift[t] = (b1[idx] - m1[idx]) * sc + be1[idx];
  } else if (slot < 8) {
    int idx = (slot - 4) * H + j;
    float sc = g2[idx] * rsqrtf(v2[idx] + BN_EPS);
    scale[t] = sc;
    shift[t] = (b2[idx] - m2[idx]) * sc + be2[idx];
  } else {
    scale[t] = 1.0f;
    shift[t] = lin1_b[j];
  }
}

// ---------- transpose+convert all weight matrices to bf16 once: wtb[s][j][k] = W_s[k][j] ----------
__global__ void k_prep_w(const float* __restrict__ W1, const float* __restrict__ W2,
                         const float* __restrict__ lin1_w, __bf16* __restrict__ wtb) {
  int t = blockIdx.x * blockDim.x + threadIdx.x;
  if (t >= 9 * H * H) return;
  int s = t >> 14;          // /(H*H)
  int rem = t & (H * H - 1);
  int j = rem >> 7, k = rem & (H - 1);
  const float* src = (s < 4) ? (W1 + (size_t)s * H * H)
                   : (s < 8) ? (W2 + (size_t)(s - 4) * H * H)
                             : lin1_w;
  wtb[(size_t)s * H * H + j * H + k] = (__bf16)src[k * H + j];
}

// ---------- argmax over atom types + embedding lookup; wave32 per node ----------
__global__ void k_embed(const float* __restrict__ vxattr, const float* __restrict__ v_embed,
                        float* __restrict__ vx, float* __restrict__ x) {
  int wave = threadIdx.x >> 5, lane = threadIdx.x & 31;
  int n = blockIdx.x * 8 + wave;
  if (n >= N_NODES) return;
  float best = -3.4e38f; int bi = 0x7fffffff;
  for (int a = lane; a < ATYPES; a += 32) {
    float v = vxattr[(long)n * ATYPES + a];
    if (v > best) { best = v; bi = a; }
  }
  for (int off = 16; off > 0; off >>= 1) {
    float ov = __shfl_xor(best, off, 32);
    int   oi = __shfl_xor(bi,   off, 32);
    if (ov > best || (ov == best && oi < bi)) { best = ov; bi = oi; }
  }
  float4 e = ((const float4*)(v_embed + (long)bi * H))[lane];
  ((float4*)(vx + (long)n * H))[lane] = e;
  ((float4*)(x  + (long)n * H))[lane] = e;
}

// ---------- agg = (1+eps[l]) * x  (layer 0 only; later layers fused into GEMM2 epilogue) ----------
__global__ void k_init_agg(float* __restrict__ agg, const float* __restrict__ x,
                           const float* __restrict__ eps, int l) {
  float s = 1.0f + eps[l];
  long i = (long)blockIdx.x * blockDim.x + threadIdx.x;
  if (i < (long)N_NODES * H / 4) {
    float4 v = ((const float4*)x)[i];
    v.x *= s; v.y *= s; v.z *= s; v.w *= s;
    ((float4*)agg)[i] = v;
  }
}

// ---------- message + scatter: agg[dst] += relu(x[src] + vx[src] + vx[dst]); wave32 per edge ----------
__global__ void k_scatter(float* __restrict__ agg, const float* __restrict__ x,
                          const float* __restrict__ vx, const int* __restrict__ ei) {
  int wave = threadIdx.x >> 5, lane = threadIdx.x & 31;
  long e = (long)blockIdx.x * 8 + wave;
  if (e >= N_EDGES) return;
  int s = ei[e];
  int d = ei[N_EDGES + e];
  float4 xa = ((const float4*)(x  + (long)s * H))[lane];
  float4 va = ((const float4*)(vx + (long)s * H))[lane];
  float4 vb = ((const float4*)(vx + (long)d * H))[lane];
  float4 m;
  m.x = fmaxf(xa.x + va.x + vb.x, 0.f);
  m.y = fmaxf(xa.y + va.y + vb.y, 0.f);
  m.z = fmaxf(xa.z + va.z + vb.z, 0.f);
  m.w = fmaxf(xa.w + va.w + vb.w, 0.f);
  float* dst = agg + (long)d * H + lane * 4;
  __hip_atomic_fetch_add(dst + 0, m.x, __ATOMIC_RELAXED, __HIP_MEMORY_SCOPE_AGENT);
  __hip_atomic_fetch_add(dst + 1, m.y, __ATOMIC_RELAXED, __HIP_MEMORY_SCOPE_AGENT);
  __hip_atomic_fetch_add(dst + 2, m.z, __ATOMIC_RELAXED, __HIP_MEMORY_SCOPE_AGENT);
  __hip_atomic_fetch_add(dst + 3, m.w, __ATOMIC_RELAXED, __HIP_MEMORY_SCOPE_AGENT);
}

// ---------- WMMA GEMM: out[M,128] = act(scale*(in @ W) + shift); W pre-transposed bf16, TDM-staged ----------
#define PW 136   // LDS pitch in bf16 (128 + 8 pad = 272B rows): conflict-free, 16B aligned; TDM pad makes it

__global__ __launch_bounds__(256) void k_gemm(const float* __restrict__ in, const __bf16* __restrict__ wt,
                                              const float* __restrict__ scale, const float* __restrict__ shift,
                                              float* __restrict__ out, float* __restrict__ out2,
                                              const float* __restrict__ eps, int eps_idx,
                                              int M, int do_relu) {
  __shared__ __attribute__((aligned(16))) __bf16 sW[H * PW];  // sW[n*PW + k] (transposed W)
  int t = threadIdx.x;
  int wave = t >> 5, lane = t & 31;
  int half = (lane >> 4) & 1;
  int l16 = lane & 15;
  int m0 = blockIdx.x * 128;

  // ---- stage W tile (128x128 bf16, transposed) via Tensor Data Mover, one issue per block ----
  // D# per CDNA5 ISA ch.8: 2D tile, data_size=8B, tile_dim0=32 (256B rows), tile_dim1=128 rows,
  // pad_enable: +16B (pad_amount=3) after every 256B (pad_interval=5) -> LDS pitch 272B = PW*2.
  if (wave == 0) {
    unsigned long long ga = (unsigned long long)(uintptr_t)(const void*)wt;
    u32 lds0 = (u32)(uintptr_t)(void*)&sW[0];
    v4u g0;
    g0[0] = 1u;                                        // count=1, user descriptor
    g0[1] = lds0;                                      // lds_addr
    g0[2] = (u32)(ga & 0xFFFFFFFFu);                   // global_addr[31:0]
    g0[3] = (u32)((ga >> 32) & 0x01FFFFFFu) | (2u << 30);  // global_addr[56:32] | type=2
    v8i g1;
    g1[0] = (int)((3u << 16) | (1u << 20) | (5u << 22) | (3u << 25)); // data_size=8B, pad_en, intv, amt
    g1[1] = (int)(32u << 16);    // tensor_dim0[15:0]=32 in bits[63:48]
    g1[2] = (int)(128u << 16);   // tensor_dim0 hi=0; tensor_dim1[15:0]=128 in bits[31:16]
    g1[3] = (int)(32u << 16);    // tensor_dim1 hi=0; tile_dim0=32 in bits[127:112]
    g1[4] = (int)128;            // tile_dim1=128; tile_dim2=0
    g1[5] = (int)32;             // tensor_dim0_stride[31:0]=32
    g1[6] = (int)(32u << 16);    // stride0 hi=0; tensor_dim1_stride[15:0]=32
    g1[7] = 0;                   // stride1 hi=0
    v4i gz = {0, 0, 0, 0};
    v8i gz8 = {0, 0, 0, 0, 0, 0, 0, 0};
    __builtin_amdgcn_tensor_load_to_lds(g0, g1, gz, gz, gz8, 0);
    __builtin_amdgcn_s_wait_tensorcnt(0);
  }
  __syncthreads();

  // ---- A fragments straight from global fp32 (rows are wave-private; L2-resident) ----
  int rowA = m0 + wave * 16 + l16;
  const float* arow = in + (long)(rowA < M ? rowA : 0) * H;  // clamp: garbage rows never stored

  v8f zero = {0.f, 0.f, 0.f, 0.f, 0.f, 0.f, 0.f, 0.f};
  v8f acc[8];
#pragma unroll
  for (int tt = 0; tt < 8; ++tt) acc[tt] = zero;

  for (int k0 = 0; k0 < H; k0 += 32) {
    // A layout: lanes 0-15 -> K {k0..k0+15}; lanes 16-31 -> +16
    int base = k0 + half * 16;
    float4 a0 = *(const float4*)(arow + base);
    float4 a1 = *(const float4*)(arow + base + 4);
    float4 a2 = *(const float4*)(arow + base + 8);
    float4 a3 = *(const float4*)(arow + base + 12);
    v16bf af;
    af[0]  = (__bf16)a0.x; af[1]  = (__bf16)a0.y; af[2]  = (__bf16)a0.z; af[3]  = (__bf16)a0.w;
    af[4]  = (__bf16)a1.x; af[5]  = (__bf16)a1.y; af[6]  = (__bf16)a1.z; af[7]  = (__bf16)a1.w;
    af[8]  = (__bf16)a2.x; af[9]  = (__bf16)a2.y; af[10] = (__bf16)a2.z; af[11] = (__bf16)a2.w;
    af[12] = (__bf16)a3.x; af[13] = (__bf16)a3.y; af[14] = (__bf16)a3.z; af[15] = (__bf16)a3.w;
    int kb = k0 + half * 16;
#pragma unroll
    for (int tt = 0; tt < 8; ++tt) {
      int n = tt * 16 + l16;
      v16bf bfrag = *(const v16bf*)(&sW[n * PW + kb]);  // 16 contiguous K of output column n
      acc[tt] = __builtin_amdgcn_wmma_f32_16x16x32_bf16(
          false, af, false, bfrag, (short)0, acc[tt], false, false);
    }
  }

  // ---- epilogue: D layout — lane holds col (l&15), rows r + 8*half; optional fused next-layer init ----
  bool wr2 = (out2 != nullptr);
  float s2 = 0.f;
  if (wr2) s2 = 1.0f + eps[eps_idx];
#pragma unroll
  for (int tt = 0; tt < 8; ++tt) {
    int n = tt * 16 + l16;
    float sc = scale[n], sh = shift[n];
#pragma unroll
    for (int r = 0; r < 8; ++r) {
      int row = m0 + wave * 16 + r + 8 * half;
      if (row < M) {
        float v = acc[tt][r] * sc + sh;
        if (do_relu) v = fmaxf(v, 0.f);
        out[(long)row * H + n] = v;
        if (wr2) out2[(long)row * H + n] = v * s2;
      }
    }
  }
}

// ---------- graph pooling ----------
__global__ void k_zero(float* p, int n) {
  int i = blockIdx.x * blockDim.x + threadIdx.x;
  if (i < n) p[i] = 0.f;
}

__global__ void k_pool(float* __restrict__ g, const float* __restrict__ x, const int* __restrict__ batch) {
  int wave = threadIdx.x >> 5, lane = threadIdx.x & 31;
  int n = blockIdx.x * 8 + wave;
  if (n >= N_NODES) return;
  int b = batch[n];
  float4 v = ((const float4*)(x + (long)n * H))[lane];
  float* dst = g + (long)b * H + lane * 4;
  __hip_atomic_fetch_add(dst + 0, v.x, __ATOMIC_RELAXED, __HIP_MEMORY_SCOPE_AGENT);
  __hip_atomic_fetch_add(dst + 1, v.y, __ATOMIC_RELAXED, __HIP_MEMORY_SCOPE_AGENT);
  __hip_atomic_fetch_add(dst + 2, v.z, __ATOMIC_RELAXED, __HIP_MEMORY_SCOPE_AGENT);
  __hip_atomic_fetch_add(dst + 3, v.w, __ATOMIC_RELAXED, __HIP_MEMORY_SCOPE_AGENT);
}

// ---------- final tiny linear: out[G,OUT] = g2 @ lin2_w + lin2_b ----------
__global__ void k_lin2(float* __restrict__ out, const float* __restrict__ g2,
                       const float* __restrict__ w, const float* __restrict__ b) {
  int t = blockIdx.x * blockDim.x + threadIdx.x;
  if (t >= GRAPHS * OUTDIM) return;
  int gi = t / OUTDIM, o = t % OUTDIM;
  float acc = b[o];
  for (int k = 0; k < H; ++k) acc += g2[gi * H + k] * w[k * OUTDIM + o];
  out[t] = acc;
}

extern "C" void kernel_launch(void* const* d_in, const int* in_sizes, int n_in,
                              void* d_out, int out_size, void* d_ws, size_t ws_size,
                              hipStream_t stream) {
  const float* v_x        = (const float*)d_in[0];
  const int*   edge_index = (const int*)d_in[1];
  // d_in[2] boundary_index: implied by edge_index (ex[e] = vx[src]+vx[dst]) -> unused
  const int*   batch      = (const int*)d_in[3];
  const float* v_embed    = (const float*)d_in[4];
  const float* eps        = (const float*)d_in[5];
  const float* W1         = (const float*)d_in[6];
  const float* b1         = (const float*)d_in[7];
  const float* bn1_gamma  = (const float*)d_in[8];
  const float* bn1_beta   = (const float*)d_in[9];
  const float* bn1_mean   = (const float*)d_in[10];
  const float* bn1_var    = (const float*)d_in[11];
  const float* W2         = (const float*)d_in[12];
  const float* b2         = (const float*)d_in[13];
  const float* bn2_gamma  = (const float*)d_in[14];
  const float* bn2_beta   = (const float*)d_in[15];
  const float* bn2_mean   = (const float*)d_in[16];
  const float* bn2_var    = (const float*)d_in[17];
  const float* lin1_w     = (const float*)d_in[18];
  const float* lin1_b     = (const float*)d_in[19];
  const float* lin2_w     = (const float*)d_in[20];
  const float* lin2_b     = (const float*)d_in[21];
  float* out = (float*)d_out;
  (void)in_sizes; (void)n_in; (void)out_size; (void)ws_size;

  char* ws = (char*)d_ws;
  size_t NH = (size_t)N_NODES * H * sizeof(float);
  float*  vx    = (float*)(ws);
  float*  x     = (float*)(ws + NH);
  float*  agg   = (float*)(ws + 2 * NH);
  char*   tail  = ws + 3 * NH;
  float*  g     = (float*)(tail);               // 64 KB
  float*  g2    = (float*)(tail + 65536);       // 64 KB
  float*  scale = (float*)(tail + 131072);      // 4.5 KB
  float*  shift = (float*)(tail + 139264);      // 4.5 KB
  __bf16* wtb   = (__bf16*)(tail + 147456);     // 9 * 128*128 bf16 = 288 KB (transposed weights)

  k_prep<<<(9 * H + 255) / 256, 256, 0, stream>>>(b1, bn1_gamma, bn1_beta, bn1_mean, bn1_var,
                                                  b2, bn2_gamma, bn2_beta, bn2_mean, bn2_var,
                                                  lin1_b, scale, shift);
  k_prep_w<<<(9 * H * H + 255) / 256, 256, 0, stream>>>(W1, W2, lin1_w, wtb);
  k_embed<<<(N_NODES + 7) / 8, 256, 0, stream>>>(v_x, v_embed, vx, x);
  k_init_agg<<<(N_NODES * H / 4 + 255) / 256, 256, 0, stream>>>(agg, x, eps, 0);

  const int gblk = (N_NODES + 127) / 128;
  for (int l = 0; l < LAYERS; ++l) {
    k_scatter<<<(N_EDGES + 7) / 8, 256, 0, stream>>>(agg, x, vx, edge_index);
    // GEMM1 in-place (each block only touches its own rows)
    k_gemm<<<gblk, 256, 0, stream>>>(agg, wtb + (size_t)l * H * H,
                                     scale + l * H, shift + l * H,
                                     agg, nullptr, eps, 0, N_NODES, 1);
    // GEMM2: writes new x, and fuses next layer's agg init = (1+eps[l+1]) * x_new
    float* o2 = (l < LAYERS - 1) ? agg : nullptr;
    k_gemm<<<gblk, 256, 0, stream>>>(agg, wtb + (size_t)(4 + l) * H * H,
                                     scale + (4 + l) * H, shift + (4 + l) * H,
                                     x, o2, eps, (l < LAYERS - 1) ? l + 1 : 0, N_NODES, 1);
  }

  k_zero<<<(GRAPHS * H + 255) / 256, 256, 0, stream>>>(g, GRAPHS * H);
  k_pool<<<(N_NODES + 7) / 8, 256, 0, stream>>>(g, x, batch);
  k_gemm<<<1, 256, 0, stream>>>(g, wtb + (size_t)8 * H * H, scale + 8 * H, shift + 8 * H,
                                g2, nullptr, eps, 0, GRAPHS, 1);
  k_lin2<<<(GRAPHS * OUTDIM + 255) / 256, 256, 0, stream>>>(out, g2, lin2_w, lin2_b);
}
